// FeatureAdaption_89567247990947
// MI455X (gfx1250) — compile-verified, matrix operand order checked by reading, not run on hardware
//
#include <hip/hip_runtime.h>
#include <hip/hip_bf16.h>

typedef _Float16 v16h __attribute__((ext_vector_type(16)));
typedef _Float16 v8h  __attribute__((ext_vector_type(8)));
typedef _Float16 h4   __attribute__((ext_vector_type(4)));
typedef float    v8f  __attribute__((ext_vector_type(8)));
typedef float    v4f  __attribute__((ext_vector_type(4)));

#define NB   8
#define CIN  256
#define COUT 256
#define CG   64      // channels per deform group
#define NG   4       // deform groups
#define HH   64
#define WW   64
#define KK2  9       // 3x3 taps
#define NOFF 72      // G*2*K2
#define NSH  36      // NA*4

// ---------------------------------------------------------------------------
// 1x1 conv: offset[b,o,h,w] = sum_i shape[b,i,h,w] * w_off[o,i]
// ---------------------------------------------------------------------------
__global__ __launch_bounds__(256) void offset_kernel(
    const float* __restrict__ shp, const float* __restrict__ w_off,
    float* __restrict__ off) {
  __shared__ float wsm[NOFF * NSH];
  for (int i = threadIdx.x; i < NOFF * NSH; i += blockDim.x) wsm[i] = w_off[i];
  __syncthreads();
  int pix = blockIdx.x * blockDim.x + threadIdx.x;   // 0 .. B*H*W-1
  int b  = pix >> 12;                                // H*W = 4096
  int hw = pix & 4095;
  float s[NSH];
#pragma unroll
  for (int i = 0; i < NSH; ++i) s[i] = shp[(b * NSH + i) * 4096 + hw];
  for (int o = 0; o < NOFF; ++o) {
    float acc = 0.f;
#pragma unroll
    for (int i = 0; i < NSH; ++i) acc += s[i] * wsm[o * NSH + i];
    off[(b * NOFF + o) * 4096 + hw] = acc;
  }
}

// ---------------------------------------------------------------------------
// x (B,Cin,H,W) f32 -> xt (B,G,H,W,Cg) f16  (channels-last per group)
// ---------------------------------------------------------------------------
__global__ __launch_bounds__(256) void xpose_kernel(
    const float* __restrict__ x, _Float16* __restrict__ xt) {
  int idx = blockIdx.x * blockDim.x + threadIdx.x;   // ((b*256+c)*64+y)*64+xw
  int xw = idx & 63;
  int y  = (idx >> 6) & 63;
  int c  = (idx >> 12) & 255;
  int b  = idx >> 20;
  int g  = c >> 6, cg = c & 63;
  xt[((((b * NG + g) * HH + y) * WW + xw) << 6) + cg] = (_Float16)x[idx];
}

// ---------------------------------------------------------------------------
// w_def (Cout,Cin,3,3) f32 -> wt[((g*9+k2)*256+o)*64+c] f16
// ---------------------------------------------------------------------------
__global__ __launch_bounds__(256) void wdef_kernel(
    const float* __restrict__ w_def, _Float16* __restrict__ wt) {
  int idx = blockIdx.x * blockDim.x + threadIdx.x;   // 36*256*64 = 589824
  int c   = idx & 63;
  int o   = (idx >> 6) & 255;
  int gk2 = idx >> 14;                               // 0..35
  int g = gk2 / 9, k2 = gk2 % 9;
  wt[idx] = (_Float16)w_def[(o * CIN + g * CG + c) * KK2 + k2];
}

// ---------------------------------------------------------------------------
// Implicit-GEMM deformable conv + ReLU on v_wmma_f32_16x16x32_f16.
// grid: x = H*(W/16) = 256, y = B ; block: 256 threads = 8 waves.
// Block covers 16 pixels x ALL 256 out channels: each wave carries two
// N-tiles (o and o+128, two f32 accumulators), so each staged A-tile feeds
// 32 WMMAs (8 waves x 2 chunks x 2 N-tiles) and the gather cost is amortized
// twice as well as with a 128-channel block.
// ---------------------------------------------------------------------------
__global__ __launch_bounds__(256, 2) void dcn_wmma_kernel(
    const _Float16* __restrict__ xt, const float* __restrict__ off,
    const _Float16* __restrict__ wt, float* __restrict__ out) {
  const int tid  = threadIdx.x;
  const int lane = tid & 31;
  const int wave = tid >> 5;
  const int h    = blockIdx.x >> 2;
  const int w0   = (blockIdx.x & 3) * 16;
  const int b    = blockIdx.y;
  const int oBase = wave * 16;          // second N-tile is oBase + 128

  __shared__ _Float16 aTile[16 * 64];   // 2KB: 16 pixels x full Cg slice
  __shared__ _Float16 geoW[16 * 4];     // 4 corner weights per pixel (0 if OOB)
  __shared__ int      geoB[16 * 4];     // clamped corner base offsets (halves)

  v8f acc0 = {0.f, 0.f, 0.f, 0.f, 0.f, 0.f, 0.f, 0.f};
  v8f acc1 = {0.f, 0.f, 0.f, 0.f, 0.f, 0.f, 0.f, 0.f};

  // staging roles: pixel sp = tid&15, channel quad cq = (tid>>4)*4 (0..60)
  const int sp = tid & 15;
  const int cq = (tid >> 4) << 2;

  // fragment roles (ISA 16-bit A/B layout): row/col = lane&15, K-half = lane>>4
  const int half = lane >> 4;
  const int am   = lane & 15;

  for (int g = 0; g < NG; ++g) {
    for (int k2 = 0; k2 < KK2; ++k2) {
      const int gk2 = g * KK2 + k2;

      if (tid < 16) {              // one thread per pixel: geometry -> LDS
        const int sw = w0 + tid;
        const float dy = off[((b * NOFF + gk2 * 2    ) * HH + h) * WW + sw];
        const float dx = off[((b * NOFF + gk2 * 2 + 1) * HH + h) * WW + sw];
        const float py = dy + (float)(k2 / 3 - 1) + (float)h;
        const float px = dx + (float)(k2 % 3 - 1) + (float)sw;
        const float y0f = floorf(py);
        const float x0f = floorf(px);
        const int   y0  = (int)y0f;
        const int   x0  = (int)x0f;
        const float wy  = py - y0f;
        const float wx  = px - x0f;
        const float cw[4] = {(1.f - wy) * (1.f - wx), (1.f - wy) * wx,
                             wy * (1.f - wx),         wy * wx};
        const int b4g = b * NG + g;
#pragma unroll
        for (int cor = 0; cor < 4; ++cor) {
          const int  yy = y0 + (cor >> 1);
          const int  xx = x0 + (cor & 1);
          const bool ok = (yy >= 0) & (yy < HH) & (xx >= 0) & (xx < WW);
          const int  yc = yy < 0 ? 0 : (yy > HH - 1 ? HH - 1 : yy);
          const int  xc = xx < 0 ? 0 : (xx > WW - 1 ? WW - 1 : xx);
          geoW[tid * 4 + cor] = (_Float16)(ok ? cw[cor] : 0.f);
          geoB[tid * 4 + cor] = ((b4g * HH + yc) * WW + xc) << 6;
        }
      }
      __syncthreads();

      // branch-free gather-blend: 4 corners x one b64 (4ch) load, pk_fma_f16
      {
        h4 av = {(_Float16)0.f, (_Float16)0.f, (_Float16)0.f, (_Float16)0.f};
#pragma unroll
        for (int cor = 0; cor < 4; ++cor) {
          const _Float16 wh = geoW[sp * 4 + cor];
          const h4 pv = *(const h4*)(xt + geoB[sp * 4 + cor] + cq);
          av += pv * wh;
        }
        *(h4*)(aTile + sp * 64 + cq) = av;
      }
      __syncthreads();

      // two K-chunks of 32 -> 2 N-tiles x 2 WMMAs per wave, A frags reused
      const _Float16* wbase0 = wt + ((gk2 * COUT + oBase + am) << 6);
      const _Float16* wbase1 = wbase0 + (128 << 6);
#pragma unroll
      for (int ck = 0; ck < 2; ++ck) {
        const _Float16* ap = aTile + am * 64 + ck * 32 + half * 8;
        v8h alo = *(const v8h*)ap;
        v8h ahi = *(const v8h*)(ap + 16);
        v16h a = __builtin_shufflevector(alo, ahi, 0, 1, 2, 3, 4, 5, 6, 7,
                                         8, 9, 10, 11, 12, 13, 14, 15);

        const _Float16* wb0 = wbase0 + ck * 32 + half * 8;
        v8h b0lo = *(const v8h*)wb0;
        v8h b0hi = *(const v8h*)(wb0 + 16);
        v16h bm0 = __builtin_shufflevector(b0lo, b0hi, 0, 1, 2, 3, 4, 5, 6, 7,
                                           8, 9, 10, 11, 12, 13, 14, 15);
        const _Float16* wb1 = wbase1 + ck * 32 + half * 8;
        v8h b1lo = *(const v8h*)wb1;
        v8h b1hi = *(const v8h*)(wb1 + 16);
        v16h bm1 = __builtin_shufflevector(b1lo, b1hi, 0, 1, 2, 3, 4, 5, 6, 7,
                                           8, 9, 10, 11, 12, 13, 14, 15);

        acc0 = __builtin_amdgcn_wmma_f32_16x16x32_f16(
            false, a, false, bm0, (short)0, acc0, false, false);
        acc1 = __builtin_amdgcn_wmma_f32_16x16x32_f16(
            false, a, false, bm1, (short)0, acc1, false, false);
      }
      if (gk2 < NG * KK2 - 1) {    // L2 hint for next tap's weight blocks
        __builtin_prefetch(wbase0 + (COUT << 6), 0, 3);
        __builtin_prefetch(wbase1 + (COUT << 6), 0, 3);
      }
      __syncthreads();             // protect aTile/geo before next tap
    }
  }

  // Epilogue: lane col am = out channel, VGPR r -> pixel = half*8 + r.
  const int wcol = w0 + half * 8;
  float* po0 = out + ((b * COUT + oBase + am      ) * HH + h) * WW + wcol;
  float* po1 = out + ((b * COUT + oBase + am + 128) * HH + h) * WW + wcol;
  v4f a0lo, a0hi, a1lo, a1hi;
#pragma unroll
  for (int r = 0; r < 4; ++r) {
    a0lo[r] = acc0[r]     > 0.f ? acc0[r]     : 0.f;
    a0hi[r] = acc0[r + 4] > 0.f ? acc0[r + 4] : 0.f;
    a1lo[r] = acc1[r]     > 0.f ? acc1[r]     : 0.f;
    a1hi[r] = acc1[r + 4] > 0.f ? acc1[r + 4] : 0.f;
  }
  *(v4f*)po0       = a0lo;
  *(v4f*)(po0 + 4) = a0hi;
  *(v4f*)po1       = a1lo;
  *(v4f*)(po1 + 4) = a1hi;
}

extern "C" void kernel_launch(void* const* d_in, const int* in_sizes, int n_in,
                              void* d_out, int out_size, void* d_ws, size_t ws_size,
                              hipStream_t stream) {
  (void)in_sizes; (void)n_in; (void)out_size; (void)ws_size;
  const float* x     = (const float*)d_in[0];
  const float* shp   = (const float*)d_in[1];
  const float* w_off = (const float*)d_in[2];
  const float* w_def = (const float*)d_in[3];
  float* out = (float*)d_out;

  char* ws = (char*)d_ws;
  float*    off = (float*)ws;                                   //  9,437,184 B
  _Float16* xt  = (_Float16*)(ws + 9437184);                    // 16,777,216 B
  _Float16* wt  = (_Float16*)(ws + 9437184 + 16777216);         //  1,179,648 B

  offset_kernel<<<dim3((NB * HH * WW) / 256), 256, 0, stream>>>(shp, w_off, off);
  xpose_kernel<<<dim3((NB * CIN * HH * WW) / 256), 256, 0, stream>>>(x, xt);
  wdef_kernel<<<dim3((36 * COUT * CG) / 256), 256, 0, stream>>>(w_def, wt);
  dcn_wmma_kernel<<<dim3(HH * (WW / 16), NB), 256, 0, stream>>>(xt, off, wt, out);
}
